// Seq2seq_47476568490026
// MI455X (gfx1250) — matrix operand order repaired and checked by reading, not compiled, not run on hardware
//
#include <hip/hip_runtime.h>
#include <hip/hip_bf16.h>

// Model dims (fixed by reference)
#define Bc   64
#define Tc   128
#define Ec   512
#define Hc   512
#define GHc  1024
#define Lc   4
#define FH   (4*Hc)    // 2048 gate width
#define TWOH (2*Hc)    // 1024 bi-dir concat
#define MB   4         // M-tiles per wave (register blocking; requires M % 64 == 0)

typedef _Float16 f16;
typedef __attribute__((ext_vector_type(16))) _Float16 v16h;
typedef __attribute__((ext_vector_type(8)))  _Float16 v8h;
typedef __attribute__((ext_vector_type(8)))  float    v8f;

__device__ __forceinline__ float sigf(float x) { return 1.0f / (1.0f + __expf(-x)); }

// ---------------- utility kernels ----------------

__global__ void fill_zero_u32(unsigned int* __restrict__ p, long n) {
    long i = (long)blockIdx.x * blockDim.x + threadIdx.x;
    if (i < n) p[i] = 0u;
}

__global__ void cvt_f32_f16_kernel(const float* __restrict__ s, f16* __restrict__ d, long n) {
    long i = (long)blockIdx.x * blockDim.x + threadIdx.x;
    if (i < n) d[i] = (f16)s[i];
}

// xe[t][b][e] = emb[x[b][t]][e]  (f16), X-row index = t*B+b for the batched GEMM
__global__ void embed_seq_kernel(const int* __restrict__ x, const float* __restrict__ emb,
                                 f16* __restrict__ xe) {
    long i = (long)blockIdx.x * 256 + threadIdx.x;     // over T*B*E
    int  e = (int)(i & (Ec - 1));
    long r = i >> 9;                                    // /E
    int  b = (int)(r & (Bc - 1));
    int  t = (int)(r >> 6);                             // /B
    int  tok = x[b * Tc + t];
    xe[i] = (f16)emb[(long)tok * Ec + e];
}

__global__ void embed_tok_kernel(const int* __restrict__ tok, const float* __restrict__ emb,
                                 f16* __restrict__ xi) {
    int i = blockIdx.x * 256 + threadIdx.x;             // over B*E
    int e = i & (Ec - 1);
    int b = i >> 9;
    xi[i] = (f16)emb[(long)tok[b] * Ec + e];
}

// ---------------- WMMA GEMM (M-blocked, optional fused second operand pair) ----
// D[M,N] = A[M,K] x W[N,K]^T  (+ A2[M,K2] x W2[N,K2]^T)  (+bias0[N]) (+bias1[N])
//          (+addend[M,N] f32)
// A/W f16 row-major (torch weight layout). Writes C (f32), optional Ch (f16 copy).
// Requires: M%64==0, N%128==0, K%32==0. Block = 256 thr = 8 waves.
// Each wave: one N-tile x MB=4 M-tiles; B fragment reused across the 4 WMMAs.
__global__ __launch_bounds__(256)
void wmma_gemm_kernel(const f16* __restrict__ A,  const f16* __restrict__ W,  int K,
                      const f16* __restrict__ A2, const f16* __restrict__ W2, int K2,
                      const float* __restrict__ bias0, const float* __restrict__ bias1,
                      const float* __restrict__ addend,
                      float* __restrict__ C, f16* __restrict__ Ch,
                      int M, int N) {
    int wave = threadIdx.x >> 5;
    int lane = threadIdx.x & 31;
    int tn = blockIdx.x * 8 + wave;        // N tile
    int m0 = blockIdx.y * (16 * MB);       // first M row of this wave's block

    int ml = lane & 15;
    int ka = (lane >> 4) << 4;   // A frag: lanes 0-15 -> K 0..15        ; 16-31 -> 16..31
    int kb = (lane >> 4) << 4;   // B frag: lanes 0-15 -> K 0..15        ; 16-31 -> 16..31
    int nrow = tn * 16 + ml;

    v8f acc[MB] = {};

    // ---- operand pair 1 ----
    {
        const f16* __restrict__ Wrow = W + (long)nrow * K;
        const f16* Ar[MB];
#pragma unroll
        for (int i = 0; i < MB; ++i) Ar[i] = A + (long)(m0 + i * 16 + ml) * K;
        for (int k0 = 0; k0 < K; k0 += 32) {
            union { v16h v; v8h h[2]; } b;
            b.h[0] = *(const v8h*)(Wrow + k0 + kb);
            b.h[1] = *(const v8h*)(Wrow + k0 + kb + 8);
#pragma unroll
            for (int i = 0; i < MB; ++i) {
                union { v16h v; v8h h[2]; } a;
                a.h[0] = *(const v8h*)(Ar[i] + k0 + ka);
                a.h[1] = *(const v8h*)(Ar[i] + k0 + ka + 8);
                acc[i] = __builtin_amdgcn_wmma_f32_16x16x32_f16(
                    false, a.v, false, b.v, (short)0, acc[i], false, false);
            }
        }
    }
    // ---- optional fused operand pair 2 (e.g. recurrent h @ Whh^T) ----
    if (A2) {
        const f16* __restrict__ Wrow = W2 + (long)nrow * K2;
        const f16* Ar[MB];
#pragma unroll
        for (int i = 0; i < MB; ++i) Ar[i] = A2 + (long)(m0 + i * 16 + ml) * K2;
        for (int k0 = 0; k0 < K2; k0 += 32) {
            union { v16h v; v8h h[2]; } b;
            b.h[0] = *(const v8h*)(Wrow + k0 + kb);
            b.h[1] = *(const v8h*)(Wrow + k0 + kb + 8);
#pragma unroll
            for (int i = 0; i < MB; ++i) {
                union { v16h v; v8h h[2]; } a;
                a.h[0] = *(const v8h*)(Ar[i] + k0 + ka);
                a.h[1] = *(const v8h*)(Ar[i] + k0 + ka + 8);
                acc[i] = __builtin_amdgcn_wmma_f32_16x16x32_f16(
                    false, a.v, false, b.v, (short)0, acc[i], false, false);
            }
        }
    }

    // C/D layout: VGPR j, lane L: M = j + 8*(L>>4), N = L&15
    int nc = tn * 16 + ml;
    float badd = 0.0f;
    if (bias0) badd += bias0[nc];
    if (bias1) badd += bias1[nc];
#pragma unroll
    for (int i = 0; i < MB; ++i) {
        int mbase = m0 + i * 16 + ((lane >> 4) << 3);
#pragma unroll
        for (int j = 0; j < 8; ++j) {
            long off = (long)(mbase + j) * N + nc;
            float v = acc[i][j] + badd;
            if (addend) v += addend[off];
            C[off] = v;
            if (Ch) Ch[off] = (f16)v;
        }
    }
}

// ---------------- LSTM gate kernels ----------------
// g[B,4H] (i,f,g,o torch order). Encoder: packed-seq masking + writes seq slice.
__global__ void lstm_gates_enc_kernel(const float* __restrict__ g, float* __restrict__ c,
                                      f16* __restrict__ h16, f16* __restrict__ seq,
                                      const int* __restrict__ xlen, int t, int dirOff) {
    int i = blockIdx.x * 256 + threadIdx.x;   // over B*H
    int b = i >> 9;
    int j = i & (Hc - 1);
    const float* gr = g + (long)b * FH;
    float ig = sigf(gr[j]);
    float fg = sigf(gr[Hc + j]);
    float gg = tanhf(gr[2 * Hc + j]);
    float og = sigf(gr[3 * Hc + j]);
    float cn = fg * c[i] + ig * gg;
    float hn = og * tanhf(cn);
    bool m = t < xlen[b];
    if (m) { c[i] = cn; h16[i] = (f16)hn; }
    seq[((long)t * Bc + b) * TWOH + dirOff + j] = m ? (f16)hn : (f16)0.0f;
}

__global__ void lstm_gates_dec_kernel(const float* __restrict__ g, float* __restrict__ c,
                                      f16* __restrict__ h16, f16* __restrict__ dnxt,
                                      int dirOff) {
    int i = blockIdx.x * 256 + threadIdx.x;   // over B*H
    int b = i >> 9;
    int j = i & (Hc - 1);
    const float* gr = g + (long)b * FH;
    float ig = sigf(gr[j]);
    float fg = sigf(gr[Hc + j]);
    float gg = tanhf(gr[2 * Hc + j]);
    float og = sigf(gr[3 * Hc + j]);
    float cn = fg * c[i] + ig * gg;
    float hn = og * tanhf(cn);
    c[i] = cn;
    h16[i] = (f16)hn;
    dnxt[(long)b * TWOH + dirOff + j] = (f16)hn;
}

// grade[b][t][j] = z3[b,:] . grW[j,:] + grb[j]   (N=2 -> scalar reduction kernel)
__global__ void grade_kernel(const float* __restrict__ z, const float* __restrict__ grW,
                             const float* __restrict__ grb, float* __restrict__ out, int t) {
    int b = blockIdx.x, j = blockIdx.y;
    __shared__ float red[128];
    float s = 0.0f;
    for (int k = threadIdx.x; k < GHc; k += 128)
        s += z[(long)b * GHc + k] * grW[(long)j * GHc + k];
    red[threadIdx.x] = s;
    __syncthreads();
    for (int st = 64; st > 0; st >>= 1) {
        if (threadIdx.x < st) red[threadIdx.x] += red[threadIdx.x + st];
        __syncthreads();
    }
    if (threadIdx.x == 0)
        out[((long)b * Tc + t) * 2 + j] = red[0] + grb[j];
}

// ---------------- host orchestration ----------------

extern "C" void kernel_launch(void* const* d_in, const int* in_sizes, int n_in,
                              void* d_out, int out_size, void* d_ws, size_t ws_size,
                              hipStream_t stream) {
    (void)in_sizes; (void)n_in; (void)out_size; (void)ws_size;

    // Input flattening: x, x_len, init_tok, embedding,
    // enc 4 layers x (Wif,Whf,bif,bhf,Wib,Whb,bib,bhb) = [4..35],
    // dec = [36..67], out_W=68, out_b=69, grader (g1W,g1b,g2W,g2b,g3W,g3b,grW,grb)=[70..77]
    const int*   x    = (const int*)d_in[0];
    const int*   xlen = (const int*)d_in[1];
    const int*   itok = (const int*)d_in[2];
    const float* emb  = (const float*)d_in[3];
    const float* encP[Lc][8];
    const float* decP[Lc][8];
    for (int l = 0; l < Lc; ++l)
        for (int i = 0; i < 8; ++i) {
            encP[l][i] = (const float*)d_in[4  + 8 * l + i];
            decP[l][i] = (const float*)d_in[36 + 8 * l + i];
        }
    const float* outW = (const float*)d_in[68];
    const float* outb = (const float*)d_in[69];
    const float* g1W = (const float*)d_in[70]; const float* g1b = (const float*)d_in[71];
    const float* g2W = (const float*)d_in[72]; const float* g2b = (const float*)d_in[73];
    const float* g3W = (const float*)d_in[74]; const float* g3b = (const float*)d_in[75];
    const float* grW = (const float*)d_in[76]; const float* grb = (const float*)d_in[77];

    // bump allocator over workspace
    char* wsb = (char*)d_ws;
    size_t off = 0;
    auto alloc = [&](size_t bytes) -> void* {
        void* p = wsb + off;
        off = (off + bytes + 255) & ~(size_t)255;
        return p;
    };

    // f16 weight arenas (L2-resident working set, ~94 MB total)
    f16 *encWi[Lc][2], *encWh[Lc][2], *decWi[Lc][2], *decWh[Lc][2];
    for (int l = 0; l < Lc; ++l)
        for (int d = 0; d < 2; ++d) {
            size_t din = (l == 0) ? Ec : TWOH;
            encWi[l][d] = (f16*)alloc((size_t)FH * din * 2);
            encWh[l][d] = (f16*)alloc((size_t)FH * Hc * 2);
            decWi[l][d] = (f16*)alloc((size_t)FH * din * 2);
            decWh[l][d] = (f16*)alloc((size_t)FH * Hc * 2);
        }
    f16* outWh = (f16*)alloc((size_t)Ec * TWOH * 2);
    f16* g1Wh  = (f16*)alloc((size_t)GHc * Ec * 2);
    f16* g2Wh  = (f16*)alloc((size_t)GHc * GHc * 2);
    f16* g3Wh  = (f16*)alloc((size_t)GHc * GHc * 2);

    // activations
    f16*   xe      = (f16*)alloc((size_t)Tc * Bc * Ec * 2);
    f16*   inp[2]  = { (f16*)alloc((size_t)Tc * Bc * TWOH * 2),
                       (f16*)alloc((size_t)Tc * Bc * TWOH * 2) };
    float* gx      = (float*)alloc((size_t)Tc * Bc * FH * 4);   // batched input projections
    float* gbuf    = (float*)alloc((size_t)Bc * FH * 4);
    f16*   hstate  = (f16*)alloc((size_t)Lc * 2 * Bc * Hc * 2); // per (layer,dir) h (f16)
    float* cstate  = (float*)alloc((size_t)Lc * 2 * Bc * Hc * 4);
    f16*   xi      = (f16*)alloc((size_t)Bc * Ec * 2);          // decoder input (fed back)
    f16*   dinp[2] = { (f16*)alloc((size_t)Bc * TWOH * 2),
                       (f16*)alloc((size_t)Bc * TWOH * 2) };
    float* obuf    = (float*)alloc((size_t)Bc * Ec * 4);
    float* z1      = (float*)alloc((size_t)Bc * GHc * 4);
    f16*   z1h     = (f16*)alloc((size_t)Bc * GHc * 2);
    float* z2      = (float*)alloc((size_t)Bc * GHc * 4);
    f16*   z2h     = (f16*)alloc((size_t)Bc * GHc * 2);
    float* z3      = (float*)alloc((size_t)Bc * GHc * 4);

    auto cvt = [&](const float* s, f16* d, long n) {
        cvt_f32_f16_kernel<<<(unsigned)((n + 255) / 256), 256, 0, stream>>>(s, d, n);
    };
    for (int l = 0; l < Lc; ++l)
        for (int d = 0; d < 2; ++d) {
            long din = (l == 0) ? Ec : TWOH;
            cvt(encP[l][d * 4 + 0], encWi[l][d], (long)FH * din);
            cvt(encP[l][d * 4 + 1], encWh[l][d], (long)FH * Hc);
            cvt(decP[l][d * 4 + 0], decWi[l][d], (long)FH * din);
            cvt(decP[l][d * 4 + 1], decWh[l][d], (long)FH * Hc);
        }
    cvt(outW, outWh, (long)Ec * TWOH);
    cvt(g1W, g1Wh, (long)GHc * Ec);
    cvt(g2W, g2Wh, (long)GHc * GHc);
    cvt(g3W, g3Wh, (long)GHc * GHc);

    auto gemm = [&](const f16* Aa, const f16* Ww, int K,
                    const f16* A2, const f16* W2, int K2,
                    const float* b0, const float* b1, const float* add,
                    float* C, f16* Ch, int M, int N) {
        dim3 grid(N / 128, M / (16 * MB));
        wmma_gemm_kernel<<<grid, 256, 0, stream>>>(Aa, Ww, K, A2, W2, K2,
                                                   b0, b1, add, C, Ch, M, N);
    };

    // ---- embedding + zero states ----
    embed_seq_kernel<<<(Tc * Bc * Ec) / 256, 256, 0, stream>>>(x, emb, xe);
    {
        long nh = (long)Lc * 2 * Bc * Hc * 2 / 4;   // u32 words of h16 arena
        long nc = (long)Lc * 2 * Bc * Hc;           // u32 words of c arena
        fill_zero_u32<<<(unsigned)((nh + 255) / 256), 256, 0, stream>>>((unsigned int*)hstate, nh);
        fill_zero_u32<<<(unsigned)((nc + 255) / 256), 256, 0, stream>>>((unsigned int*)cstate, nc);
    }

    // ---- encoder: per layer/dir, batched x-projection then sequential recurrence ----
    const f16* curin = xe;
    int curK = Ec;
    for (int l = 0; l < Lc; ++l) {
        f16* nxt = inp[l & 1];
        for (int d = 0; d < 2; ++d) {
            const float* bi = encP[l][d * 4 + 2];
            const float* bh = encP[l][d * 4 + 3];
            // gx[T*B, 4H] = X @ Wih^T + (bih + bhh)   (fully parallel)
            gemm(curin, encWi[l][d], curK, nullptr, nullptr, 0,
                 bi, bh, nullptr, gx, nullptr, Tc * Bc, FH);
            f16*   hst = hstate + (size_t)(l * 2 + d) * Bc * Hc;
            float* cst = cstate + (size_t)(l * 2 + d) * Bc * Hc;
            for (int s = 0; s < Tc; ++s) {
                int t = d ? (Tc - 1 - s) : s;
                // g = h @ Whh^T + gx[t]
                gemm(hst, encWh[l][d], Hc, nullptr, nullptr, 0,
                     nullptr, nullptr, gx + (size_t)t * Bc * FH, gbuf, nullptr, Bc, FH);
                lstm_gates_enc_kernel<<<(Bc * Hc) / 256, 256, 0, stream>>>(
                    gbuf, cst, hst, nxt, xlen, t, d * Hc);
            }
        }
        curin = nxt;
        curK = TWOH;
    }

    // ---- decoder: autoregressive, states continue from encoder finals ----
    embed_tok_kernel<<<(Bc * Ec) / 256, 256, 0, stream>>>(itok, emb, xi);
    for (int t = 0; t < Tc; ++t) {
        const f16* din = xi;
        int dK = Ec;
        for (int l = 0; l < Lc; ++l) {
            f16* dnxt = dinp[l & 1];
            for (int d = 0; d < 2; ++d) {
                const float* bi  = decP[l][d * 4 + 2];
                const float* bh  = decP[l][d * 4 + 3];
                f16*   hst = hstate + (size_t)(l * 2 + d) * Bc * Hc;
                float* cst = cstate + (size_t)(l * 2 + d) * Bc * Hc;
                // fused: g = x @ Wih^T + h @ Whh^T + bih + bhh   (one kernel)
                gemm(din, decWi[l][d], dK, hst, decWh[l][d], Hc,
                     bi, bh, nullptr, gbuf, nullptr, Bc, FH);
                lstm_gates_dec_kernel<<<(Bc * Hc) / 256, 256, 0, stream>>>(
                    gbuf, cst, hst, dnxt, d * Hc);
            }
            din = dnxt;
            dK = TWOH;
        }
        // out = concat @ out_W^T + out_b  (f16 copy into xi = next step's input)
        gemm(din, outWh, TWOH, nullptr, nullptr, 0,
             outb, nullptr, nullptr, obuf, xi, Bc, Ec);
        // grader MLP
        gemm(xi,  g1Wh, Ec,  nullptr, nullptr, 0, g1b, nullptr, nullptr, z1, z1h, Bc, GHc);
        gemm(z1h, g2Wh, GHc, nullptr, nullptr, 0, g2b, nullptr, nullptr, z2, z2h, Bc, GHc);
        gemm(z2h, g3Wh, GHc, nullptr, nullptr, 0, g3b, nullptr, nullptr, z3, nullptr, Bc, GHc);
        grade_kernel<<<dim3(Bc, 2), 128, 0, stream>>>(z3, grW, grb, (float*)d_out, t);
    }
}
